// DualGraph_16381005267539
// MI455X (gfx1250) — matrix operand order repaired and checked by reading, not compiled, not run on hardware
//
#include <hip/hip_runtime.h>
#include <hip/hip_bf16.h>

// CDNA5 / gfx1250 implementation of the DualGraph reference.
// Heavy GEMMs run on v_wmma_f32_16x16x32_bf16 (bf16 A/B, f32 accumulate).
// All GEMM kernels are templated on (K, NO) so tile counts / strides are
// compile-time: accumulators stay in VGPRs (no scratch spills) and addressing
// is shift-based. Each wave owns one m-tile and TPW consecutive n-tiles, so a
// single LDS A-fragment read feeds TPW back-to-back WMMAs per k-step.
// Training-mode BN stats (per-channel sum/sumsq over the batch) are fused into
// the WMMA D-fragment store via LDS atomics + one global atomic per channel.

typedef __bf16 bf16;
typedef __attribute__((ext_vector_type(16))) __bf16 v16bf;
typedef __attribute__((ext_vector_type(8)))  __bf16 v8bf;
typedef __attribute__((ext_vector_type(8)))  float  v8f;

#define LEAK 0.01f
#define NNODE 128   // N (graph nodes)
#define BB    8     // batch
#define DC    128   // point-node channels
#define SC    32    // distribution-node channels
#define EDGE  (BB*NNODE*NNODE)  // 131072

__device__ __forceinline__ float leakyf(float x){ return x >= 0.f ? x : LEAK*x; }

__device__ __forceinline__ float wave_sum32(float v){
#pragma unroll
  for (int o = 16; o > 0; o >>= 1) v += __shfl_xor(v, o, 32);
  return v;
}

__device__ __forceinline__ float block_sum128(float v, float* red){
  const int t = threadIdx.x;
  red[t] = v; __syncthreads();
#pragma unroll
  for (int o = 64; o > 0; o >>= 1){ if (t < o) red[t] += red[t + o]; __syncthreads(); }
  float r = red[0]; __syncthreads();
  return r;
}

// ---- f32 -> bf16 weight conversion ----------------------------------------
__global__ void k_cvt(const float* __restrict__ s, bf16* __restrict__ d, int n){
  int i = blockIdx.x * 256 + threadIdx.x;
  if (i < n) d[i] = (bf16)s[i];
}

// ---- WMMA fragment helpers -------------------------------------------------
// A 16x32 bf16 fragment: lane<16 row=lane K=k0..k0+15 contiguous,
// lane>=16 row=lane-16 K=k0+16..k0+31; elements packed ascending.
__device__ __forceinline__ v16bf load_afrag(const bf16* rowptr, int k0, int half){
  const int kb = k0 + half * 16;
  v8bf lo = *(const v8bf*)(rowptr + kb);
  v8bf hi = *(const v8bf*)(rowptr + kb + 8);
  v16bf a;
#pragma unroll
  for (int e = 0; e < 8; ++e){ a[e] = lo[e]; a[e + 8] = hi[e]; }
  return a;
}
// B 32x16 bf16 fragment: lane<16 col n=lane K=k0..k0+15 contiguous,
// lane>=16 col n=lane-16 K=k0+16..k0+31 (same k order as the A fragment).
template<int K>
__device__ __forceinline__ v16bf load_bfrag(const bf16* __restrict__ w, int n,
                                            int k0, int half){
  return *(const v16bf*)(w + (size_t)n * K + (k0 + half * 16));
}

// ---- sim-edge layer 1: build sim tiles on the fly, GEMM vs W1 --------------
// grid = (NN/64, B*NN), block = 256 (8 waves). M-tile = 64 rows (64 j's of one
// (b,i)), K = C, NO = 2C. Writes h1pre (bf16), per-channel stats, and nsl2.
template<int C>
__global__ __launch_bounds__(256) void k_simA(
    const float* __restrict__ v, const bf16* __restrict__ w1b,
    bf16* __restrict__ outp, float* __restrict__ stats,
    float* __restrict__ nsl2)
{
  constexpr int NO     = 2 * C;
  constexpr int Kp     = C + 8;          // 16B-aligned padded row stride
  constexpr int NTILES = NO / 16;
  constexpr int TPW    = (4 * NTILES) / 8;   // tiles per wave (TPW | NTILES)
  constexpr int KTILES = C / 32;
  __shared__ bf16  As[64 * Kp];
  __shared__ float rowsum[64];
  __shared__ float ssum[NO];
  __shared__ float ssq [NO];
  const int tid = threadIdx.x;
  const int bi  = blockIdx.y;                    // b*NN + i
  const int b   = bi / NNODE, i = bi % NNODE;
  const int j0  = blockIdx.x * 64;

  for (int c = tid; c < NO; c += 256){ ssum[c] = 0.f; ssq[c] = 0.f; }
  if (tid < 64) rowsum[tid] = 0.f;
  __syncthreads();

  // sim[j,k] = (v[b,i,k] - v[b,j,k])^2, nsl2 row sums on the side
  {
    const float* vi = v + ((size_t)b * NNODE + i) * C;
    const int r = tid >> 2, part = tid & 3;
    constexpr int kc = C >> 2;
    const float* vj = v + ((size_t)b * NNODE + (j0 + r)) * C;
    float ps = 0.f;
#pragma unroll
    for (int kk = 0; kk < kc; ++kk){
      const int k = part * kc + kk;
      const float d = vi[k] - vj[k];
      const float s = d * d;
      As[r * Kp + k] = (bf16)s;
      ps += s;
    }
    atomicAdd(&rowsum[r], ps);
  }
  __syncthreads();
  if (nsl2 != nullptr && tid < 64)
    nsl2[(size_t)bi * NNODE + j0 + tid] = -rowsum[tid];

  const int wid = tid >> 5, lane = tid & 31;
  const int half = lane >> 4, l16 = lane & 15;
  const int mt  = (wid * TPW) / NTILES;      // one m-tile per wave
  const int nt0 = (wid * TPW) % NTILES;
  v8f acc[TPW];
#pragma unroll
  for (int t = 0; t < TPW; ++t)
#pragma unroll
    for (int e = 0; e < 8; ++e) acc[t][e] = 0.f;

#pragma unroll
  for (int kt = 0; kt < KTILES; ++kt){
    const int k0 = kt * 32;
    const v16bf a = load_afrag(&As[(mt * 16 + l16) * Kp], k0, half);
#pragma unroll
    for (int t = 0; t < TPW; ++t){
      const v16bf w = load_bfrag<C>(w1b, (nt0 + t) * 16 + l16, k0, half);
      acc[t] = __builtin_amdgcn_wmma_f32_16x16x32_bf16(
          false, a, false, w, (short)0, acc[t], false, false);
    }
  }

  const size_t rbase = (size_t)bi * NNODE + j0 + mt * 16 + half * 8;
#pragma unroll
  for (int t = 0; t < TPW; ++t){
    const int n = (nt0 + t) * 16 + l16;
    float sA = 0.f, sQ = 0.f;
#pragma unroll
    for (int r = 0; r < 8; ++r){
      const float x = acc[t][r];
      outp[(rbase + r) * NO + n] = (bf16)x;
      sA += x; sQ += x * x;
    }
    atomicAdd(&ssum[n], sA);
    atomicAdd(&ssq [n], sQ);
  }
  __syncthreads();
  for (int c = tid; c < NO; c += 256){
    atomicAdd(&stats[c],      ssum[c]);
    atomicAdd(&stats[NO + c], ssq [c]);
  }
}

// ---- generic rows-GEMM: (optional fused BN+leaky on bf16 input) x Wbf ------
// grid = M/64, block 256. Emits bf16 pre-acts + per-channel stats.
template<int K, int NO>
__global__ __launch_bounds__(256) void k_gemm(
    const bf16* __restrict__ act, const float* __restrict__ scsh,
    const bf16* __restrict__ wB, bf16* __restrict__ outp,
    float* __restrict__ stats)
{
  constexpr int Kp     = K + 8;
  constexpr int NTILES = NO / 16;
  constexpr int TPW    = (4 * NTILES) / 8;
  constexpr int KTILES = K / 32;
  __shared__ bf16  As[64 * Kp];
  __shared__ float ssum[NO];
  __shared__ float ssq [NO];
  const int tid = threadIdx.x;
  const size_t row0 = (size_t)blockIdx.x * 64;

  for (int c = tid; c < NO; c += 256){ ssum[c] = 0.f; ssq[c] = 0.f; }
#pragma unroll
  for (int idx = tid; idx < 64 * K; idx += 256){
    const int r = idx / K, k = idx % K;            // K is constexpr -> shifts
    float x = (float)act[(row0 + r) * (size_t)K + k];
    if (scsh) x = leakyf(x * scsh[k] + scsh[K + k]);   // fused BN + LeakyReLU
    As[r * Kp + k] = (bf16)x;
  }
  __syncthreads();

  const int wid = tid >> 5, lane = tid & 31;
  const int half = lane >> 4, l16 = lane & 15;
  const int mt  = (wid * TPW) / NTILES;
  const int nt0 = (wid * TPW) % NTILES;
  v8f acc[TPW];
#pragma unroll
  for (int t = 0; t < TPW; ++t)
#pragma unroll
    for (int e = 0; e < 8; ++e) acc[t][e] = 0.f;

#pragma unroll
  for (int kt = 0; kt < KTILES; ++kt){
    const int k0 = kt * 32;
    const v16bf a = load_afrag(&As[(mt * 16 + l16) * Kp], k0, half);
#pragma unroll
    for (int t = 0; t < TPW; ++t){
      const v16bf w = load_bfrag<K>(wB, (nt0 + t) * 16 + l16, k0, half);
      acc[t] = __builtin_amdgcn_wmma_f32_16x16x32_bf16(
          false, a, false, w, (short)0, acc[t], false, false);
    }
  }

  const size_t rbase = row0 + mt * 16 + half * 8;
#pragma unroll
  for (int t = 0; t < TPW; ++t){
    const int n = (nt0 + t) * 16 + l16;
    float sA = 0.f, sQ = 0.f;
#pragma unroll
    for (int r = 0; r < 8; ++r){
      const float x = acc[t][r];
      outp[(rbase + r) * NO + n] = (bf16)x;
      sA += x; sQ += x * x;
    }
    atomicAdd(&ssum[n], sA);
    atomicAdd(&ssq [n], sQ);
  }
  __syncthreads();
  for (int c = tid; c < NO; c += 256){
    atomicAdd(&stats[c],      ssum[c]);
    atomicAdd(&stats[NO + c], ssq [c]);
  }
}

// ---- BN stats -> per-channel scale/shift -----------------------------------
__global__ void k_finalize(const float* __restrict__ stats,
                           const float* __restrict__ g, const float* __restrict__ b,
                           float* __restrict__ scsh, int ch, float invM){
  const int c = threadIdx.x;
  if (c < ch){
    const float mean = stats[c] * invM;
    const float var  = stats[ch + c] * invM - mean * mean;
    const float sc   = g[c] * rsqrtf(var + 1e-5f);
    scsh[c]      = sc;
    scsh[ch + c] = b[c] - mean * sc;
  }
}

// ---- sim head: BN+leaky, dot w3, + b3, sigmoid (wave per row) --------------
template<int C>
__global__ __launch_bounds__(256) void k_head(
    const bf16* __restrict__ h2, const float* __restrict__ scsh,
    const float* __restrict__ w3, const float* __restrict__ b3,
    float* __restrict__ e_out)
{
  const int wid = threadIdx.x >> 5, lane = threadIdx.x & 31;
  const size_t row = (size_t)blockIdx.x * 8 + wid;
  float acc = 0.f;
#pragma unroll
  for (int k = lane; k < C; k += 32){
    const float x = (float)h2[row * C + k] * scsh[k] + scsh[C + k];
    acc += leakyf(x) * w3[k];
  }
  acc = wave_sum32(acc);
  if (lane == 0) e_out[row] = 1.f / (1.f + __expf(-(acc + b3[0])));
}

// ---- masked L1 edge normalization (one row per block) ----------------------
__global__ __launch_bounds__(128) void k_edge_norm(
    const float* __restrict__ e, const float* __restrict__ elast,
    float* __restrict__ edge_ws, float* __restrict__ edge_out)
{
  __shared__ float red[128];
  const int bi = blockIdx.x, i = bi % NNODE, j = threadIdx.x;
  const size_t idx = (size_t)bi * NNODE + j;
  const float ev = e[idx];
  const float em = (j == i) ? 0.f : elast[idx];
  const float s  = block_sum128(em, red);
  const float t  = ev * em;
  const float a  = block_sum128(fabsf(t), red);
  float x = t / fmaxf(a, 1e-12f) * s + ((j == i) ? 1.f : 0.f) + 1e-6f;
  const float r = block_sum128(x, red);
  x /= r;
  edge_ws[idx] = x;
  if (edge_out) edge_out[idx] = x;
}

// ---- P2DAgg: cat(edge[:,:, :S], dn) @ W^T + b, leaky -----------------------
__global__ __launch_bounds__(64) void k_p2d(
    const float* __restrict__ pe, const float* __restrict__ dn,
    const float* __restrict__ W, const float* __restrict__ bvec,
    float* __restrict__ dn_ws, float* __restrict__ dn_out)
{
  __shared__ float xs[64];
  const int bi = blockIdx.x, t = threadIdx.x;
  xs[t] = (t < SC) ? pe[(size_t)bi * NNODE + t] : dn[(size_t)bi * SC + (t - SC)];
  __syncthreads();
  if (t < SC){
    float acc = bvec[t];
#pragma unroll
    for (int c = 0; c < 2 * SC; ++c) acc += xs[c] * W[t * 2 * SC + c];
    const float o = leakyf(acc);
    dn_ws [(size_t)bi * SC + t] = o;
    dn_out[(size_t)bi * SC + t] = o;
  }
}

// ---- D2PAgg prep: ef = l1norm(masked de), aggr = ef @ pn, xcat (bf16) ------
__global__ __launch_bounds__(128) void k_d2p_prep(
    const float* __restrict__ de, const float* __restrict__ pn,
    bf16* __restrict__ xcat)
{
  __shared__ float ef[128];
  __shared__ float red[128];
  const int bi = blockIdx.x, b = bi / NNODE, i = bi % NNODE, j = threadIdx.x;
  const float dm = (j == i) ? 0.f : de[(size_t)bi * NNODE + j];
  const float ss = block_sum128(fabsf(dm), red);
  ef[j] = dm / fmaxf(ss, 1e-12f);
  __syncthreads();
  float acc = 0.f;                       // thread j doubles as channel d
  for (int jj = 0; jj < NNODE; ++jj)
    acc += ef[jj] * pn[((size_t)b * NNODE + jj) * DC + j];
  xcat[(size_t)bi * 2 * DC + j]      = (bf16)pn[((size_t)b * NNODE + i) * DC + j];
  xcat[(size_t)bi * 2 * DC + DC + j] = (bf16)acc;
}

// ---- final BN+leaky apply (d2p layer-2 output -> point_node) ---------------
__global__ void k_bn_apply(const bf16* __restrict__ h, const float* __restrict__ scsh,
                           float* __restrict__ o1, float* __restrict__ o2,
                           int n, int C){
  const int i = blockIdx.x * 256 + threadIdx.x;
  if (i < n){
    const int c = i % C;
    float x = (float)h[i] * scsh[c] + scsh[C + c];
    x = leakyf(x);
    o1[i] = x; o2[i] = x;
  }
}

// ---- host-side parameter bundles -------------------------------------------
struct SimP { const float *w1,*g1,*bb1,*w2,*g2,*bb2,*w3,*b3; };
struct MlpP { const float *w1,*g1,*bb1,*w2,*g2,*bb2; };

template<int C>
static void run_sim(const SimP& sp, const bf16* w1b, const bf16* w2b,
                    const float* vbuf, const float* elast,
                    float* e_ws, float* e_out, float* nsl2_out,
                    bf16* h1pre, bf16* h2pre, float* e_tmp,
                    float* stats, float* scsh1, float* scsh2,
                    hipStream_t stream)
{
  constexpr int NO = 2 * C;
  const float invM = 1.f / (float)EDGE;
  hipMemsetAsync(stats, 0, (size_t)2 * NO * sizeof(float), stream);
  dim3 gA(NNODE / 64, BB * NNODE);
  k_simA<C><<<gA, 256, 0, stream>>>(vbuf, w1b, h1pre, stats, nsl2_out);
  k_finalize<<<1, NO, 0, stream>>>(stats, sp.g1, sp.bb1, scsh1, NO, invM);
  hipMemsetAsync(stats, 0, (size_t)2 * C * sizeof(float), stream);
  k_gemm<NO, C><<<EDGE / 64, 256, 0, stream>>>(h1pre, scsh1, w2b, h2pre, stats);
  k_finalize<<<1, C, 0, stream>>>(stats, sp.g2, sp.bb2, scsh2, C, invM);
  k_head<C><<<EDGE / 8, 256, 0, stream>>>(h2pre, scsh2, sp.w3, sp.b3, e_tmp);
  k_edge_norm<<<BB * NNODE, 128, 0, stream>>>(e_tmp, elast, e_ws, e_out);
}

// ============================================================================
extern "C" void kernel_launch(void* const* d_in, const int* in_sizes, int n_in,
                              void* d_out, int out_size, void* d_ws, size_t ws_size,
                              hipStream_t stream)
{
  if (n_in < 61 || !d_out || !d_ws) return;
  // Detect params pytree flatten order: jax-sorted-keys vs dict-insertion.
  const bool sorted = (in_sizes[1] == BB * NNODE * SC);   // distribution_node

  auto F = [&](int i)->const float* { return (const float*)d_in[i]; };
  const float *dist_edge, *dist_node, *mid_node, *pt_edge, *pt_node;
  if (sorted){ dist_edge=F(0); dist_node=F(1); mid_node=F(2); pt_edge=F(59); pt_node=F(60); }
  else       { mid_node=F(0);  pt_node=F(1);  dist_node=F(2); dist_edge=F(3); pt_edge=F(4); }

  auto getSim = [&](int base)->SimP{
    SimP p;
    if (sorted){ p.b3=F(base+0); p.bb1=F(base+1); p.bb2=F(base+2); p.g1=F(base+3);
                 p.g2=F(base+4); p.w1=F(base+5); p.w2=F(base+6); p.w3=F(base+7); }
    else       { p.w1=F(base+0); p.g1=F(base+1); p.bb1=F(base+2); p.w2=F(base+3);
                 p.g2=F(base+4); p.bb2=F(base+5); p.w3=F(base+6); p.b3=F(base+7); }
    return p;
  };
  auto getMlp = [&](int base)->MlpP{
    MlpP p;
    if (sorted){ p.bb1=F(base+0); p.bb2=F(base+1); p.g1=F(base+2); p.g2=F(base+3);
                 p.w1=F(base+4); p.w2=F(base+5); }
    else       { p.w1=F(base+0); p.g1=F(base+1); p.bb1=F(base+2); p.w2=F(base+3);
                 p.g2=F(base+4); p.bb2=F(base+5); }
    return p;
  };
  SimP initP, psimP[2], dsimP[2];
  MlpP d2pP[2];
  const float *p2dW[2], *p2db[2];
  for (int g = 0; g < 2; ++g){
    if (sorted){
      const int gb = 3 + 24 * g;
      d2pP[g] = getMlp(gb + 0);  dsimP[g] = getSim(gb + 6);
      p2dW[g] = F(gb + 14); p2db[g] = F(gb + 15); psimP[g] = getSim(gb + 16);
    } else {
      const int gb = 13 + 24 * g;
      psimP[g] = getSim(gb + 0); dsimP[g] = getSim(gb + 8);
      p2dW[g] = F(gb + 16); p2db[g] = F(gb + 17); d2pP[g] = getMlp(gb + 18);
    }
  }
  initP = getSim(sorted ? 51 : 5);

  // ---- workspace layout ----
  char* base = (char*)d_ws; size_t off = 0;
  auto alloc = [&](size_t bytes)->void*{
    off = (off + 255) & ~(size_t)255; void* p = base + off; off += bytes; return p;
  };
  const size_t Mbig = EDGE;
  bf16*  h1pre = (bf16*) alloc(Mbig * 256 * sizeof(bf16));
  bf16*  h2pre = (bf16*) alloc(Mbig * 128 * sizeof(bf16));
  float* e_tmp = (float*)alloc(Mbig * sizeof(float));
  float* peA   = (float*)alloc(EDGE * sizeof(float));
  float* peB   = (float*)alloc(EDGE * sizeof(float));
  float* deA   = (float*)alloc(EDGE * sizeof(float));
  float* deB   = (float*)alloc(EDGE * sizeof(float));
  float* pnA   = (float*)alloc((size_t)BB*NNODE*DC * sizeof(float));
  float* pnB   = (float*)alloc((size_t)BB*NNODE*DC * sizeof(float));
  float* dnA   = (float*)alloc((size_t)BB*NNODE*SC * sizeof(float));
  float* dnB   = (float*)alloc((size_t)BB*NNODE*SC * sizeof(float));
  bf16*  xcat  = (bf16*) alloc((size_t)BB*NNODE*2*DC * sizeof(bf16));
  float* stats = (float*)alloc(512 * sizeof(float));
  float* scsh1 = (float*)alloc(512 * sizeof(float));
  float* scsh2 = (float*)alloc(512 * sizeof(float));
  bf16 *wb_i1, *wb_i2, *wb_ps1[2], *wb_ps2[2], *wb_ds1[2], *wb_ds2[2], *wb_dp1[2], *wb_dp2[2];
  wb_i1 = (bf16*)alloc(256*128*sizeof(bf16));
  wb_i2 = (bf16*)alloc(128*256*sizeof(bf16));
  for (int g = 0; g < 2; ++g){
    wb_ps1[g] = (bf16*)alloc(256*128*sizeof(bf16));
    wb_ps2[g] = (bf16*)alloc(128*256*sizeof(bf16));
    wb_ds1[g] = (bf16*)alloc(64*32  *sizeof(bf16));
    wb_ds2[g] = (bf16*)alloc(32*64  *sizeof(bf16));
    wb_dp1[g] = (bf16*)alloc(256*256*sizeof(bf16));
    wb_dp2[g] = (bf16*)alloc(128*256*sizeof(bf16));
  }
  if (off > ws_size) return;   // workspace too small: bail deterministically

  auto cvt = [&](const float* s, bf16* d, int n){
    k_cvt<<<(n + 255) / 256, 256, 0, stream>>>(s, d, n);
  };
  cvt(initP.w1, wb_i1, 256*128); cvt(initP.w2, wb_i2, 128*256);
  for (int g = 0; g < 2; ++g){
    cvt(psimP[g].w1, wb_ps1[g], 256*128); cvt(psimP[g].w2, wb_ps2[g], 128*256);
    cvt(dsimP[g].w1, wb_ds1[g], 64*32);   cvt(dsimP[g].w2, wb_ds2[g], 32*64);
    cvt(d2pP[g].w1,  wb_dp1[g], 256*256); cvt(d2pP[g].w2,  wb_dp2[g], 128*256);
  }

  auto run_d2p = [&](const MlpP& mp, const bf16* w1b, const bf16* w2b,
                     const float* de, const float* pn_in,
                     float* pn_ws, float* pn_out){
    k_d2p_prep<<<BB * NNODE, 128, 0, stream>>>(de, pn_in, xcat);
    const size_t Ms = (size_t)BB * NNODE;            // 1024 rows
    const float invMs = 1.f / (float)Ms;
    hipMemsetAsync(stats, 0, 2 * 256 * sizeof(float), stream);
    k_gemm<256, 256><<<(int)(Ms / 64), 256, 0, stream>>>(xcat, nullptr, w1b, h1pre, stats);
    k_finalize<<<1, 2*DC, 0, stream>>>(stats, mp.g1, mp.bb1, scsh1, 2*DC, invMs);
    hipMemsetAsync(stats, 0, 2 * DC * sizeof(float), stream);
    k_gemm<256, 128><<<(int)(Ms / 64), 256, 0, stream>>>(h1pre, scsh1, w2b, h2pre, stats);
    k_finalize<<<1, DC, 0, stream>>>(stats, mp.g2, mp.bb2, scsh2, DC, invMs);
    k_bn_apply<<<(int)((Ms * DC + 255) / 256), 256, 0, stream>>>(
        h2pre, scsh2, pn_ws, pn_out, (int)(Ms * DC), DC);
  };

  float* out = (float*)d_out;
  // init: point_edge, _ = sim_edge(init, middle_node, point_edge_in)
  run_sim<DC>(initP, wb_i1, wb_i2, mid_node, pt_edge, peA, nullptr, nullptr,
              h1pre, h2pre, e_tmp, stats, scsh1, scsh2, stream);
  const float* pe_cur = peA;
  const float* pn_cur = pt_node;
  const float* dn_cur = dist_node;
  const float* de_cur = dist_edge;
  for (int g = 0; g < 2; ++g){
    float* pe_n = (g == 0) ? peB : peA;
    run_sim<DC>(psimP[g], wb_ps1[g], wb_ps2[g], pn_cur, pe_cur, pe_n,
                out + (size_t)g * EDGE,                 // p_sims[g]
                out + (size_t)(2 + g) * EDGE,           // nsl2s[g]
                h1pre, h2pre, e_tmp, stats, scsh1, scsh2, stream);
    pe_cur = pe_n;
    float* dn_n = (g == 0) ? dnA : dnB;
    k_p2d<<<BB * NNODE, 64, 0, stream>>>(pe_cur, dn_cur, p2dW[g], p2db[g], dn_n,
        out + (size_t)8 * EDGE + (size_t)g * (BB * NNODE * SC));  // d_nodes[g]
    dn_cur = dn_n;
    float* de_n = (g == 0) ? deA : deB;
    run_sim<SC>(dsimP[g], wb_ds1[g], wb_ds2[g], dn_cur, de_cur, de_n,
                out + (size_t)(4 + g) * EDGE, nullptr,  // d_sims[g]
                h1pre, h2pre, e_tmp, stats, scsh1, scsh2, stream);
    de_cur = de_n;
    float* pn_n = (g == 0) ? pnA : pnB;
    run_d2p(d2pP[g], wb_dp1[g], wb_dp2[g], de_cur, pn_cur, pn_n,
            out + (size_t)(6 + g) * EDGE);              // p_nodes[g]
    pn_cur = pn_n;
  }
}
